// CustomSAGEConv_27410481283882
// MI455X (gfx1250) — matrix operand order repaired and checked by reading, not compiled, not run on hardware
//
#include <hip/hip_runtime.h>

// Problem constants (fixed by the reference)
#define NN 10000   // nodes
#define DD 32      // in-degree / timesteps
#define FF 512     // feature dim == hidden dim
#define G4 2048    // 4*FF (gates i,f,g,o)
#define KT (FF/32) // k-tiles of 32 per GEMM

typedef __attribute__((ext_vector_type(16))) _Float16 v16h;
typedef __attribute__((ext_vector_type(8)))  _Float16 v8h;
typedef __attribute__((ext_vector_type(8)))  float    v8f;

// ---- WMMA fragment loaders (CDNA5 wave32, 16x16x32 f16) --------------------
// A matrix 16x32 f16 (ISA 7.12.2): lane<16 -> row=lane, K={0..7}U{16..23};
// lane>=16 -> row=lane-16, K={8..15}U{24..31}. Two 16B chunks per lane.
__device__ __forceinline__ v16h frag_a(const _Float16* p, int lane) {
  const _Float16* q = p + ((lane & 16) ? 8 : 0);
  v8h lo = *(const v8h*)(q);
  v8h hi = *(const v8h*)(q + 16);
  return __builtin_shufflevector(lo, hi, 0,1,2,3,4,5,6,7,8,9,10,11,12,13,14,15);
}
// B matrix 32x16 f16: lane<16 -> col=lane, K={0..7}U{16..23}; lane>=16 -> K={8..15}U{24..31}.
// Two 16B chunks per lane (b stored row-major [col][k]).
__device__ __forceinline__ v16h frag_b(const _Float16* p, int lane) {
  const _Float16* q = p + ((lane & 16) ? 8 : 0);
  v8h lo = *(const v8h*)(q);
  v8h hi = *(const v8h*)(q + 16);
  return __builtin_shufflevector(lo, hi, 0,1,2,3,4,5,6,7,8,9,10,11,12,13,14,15);
}
__device__ __forceinline__ v8f wmma16(v16h a, v16h b, v8f c) {
  return __builtin_amdgcn_wmma_f32_16x16x32_f16(false, a, false, b, (short)0, c, false, false);
}
__device__ __forceinline__ float sigm(float x) { return 1.0f / (1.0f + __expf(-x)); }

// ---- fp32 -> fp16 convert (grid-stride) ------------------------------------
__global__ void k_cvt(const float* __restrict__ in, _Float16* __restrict__ out, int n) {
  int i = blockIdx.x * blockDim.x + threadIdx.x;
  int s = gridDim.x * blockDim.x;
  for (; i < n; i += s) out[i] = (_Float16)in[i];
}

// ---- fp32 [F,F] -> fp16 transposed [j][k] = in[k][j] -----------------------
__global__ void k_tr(const float* __restrict__ in, _Float16* __restrict__ out) {
  int i = blockIdx.x * blockDim.x + threadIdx.x;
  int s = gridDim.x * blockDim.x;
  for (; i < FF * FF; i += s) {
    int j = i >> 9, k = i & (FF - 1);
    out[i] = (_Float16)in[k * FF + j];
  }
}

// ---- WMMA GEMM: C16[m,j] = A1@B1^T (+ A2@B2^T) (+ bias1 + bias2) -----------
// A* : [M,512] f16 row-major.  B* : [ncols,512] f16 row-major (k contiguous).
// Wave computes a 16x64 strip: blockIdx.x -> row tile, colBase = bIdx.y*512+wave*64.
__global__ __launch_bounds__(256) void k_gemm(
    const _Float16* __restrict__ A1, const _Float16* __restrict__ B1,
    const _Float16* __restrict__ A2, const _Float16* __restrict__ B2,
    const float* __restrict__ bias1, const float* __restrict__ bias2,
    _Float16* __restrict__ C, int ncols)
{
  const int tid = threadIdx.x, lane = tid & 31, wave = tid >> 5;
  const int rowBase = blockIdx.x << 4;
  const int colBase = (blockIdx.y << 9) + (wave << 6);
  const int jn = lane & 15;
  const int hi8 = (lane >> 4) << 3;

  v8f acc[4];
#pragma unroll
  for (int ct = 0; ct < 4; ++ct) acc[ct] = (v8f){0.f,0.f,0.f,0.f,0.f,0.f,0.f,0.f};

  const _Float16* aRow = A1 + (size_t)(rowBase + jn) * FF;
  for (int kt = 0; kt < KT; ++kt) {
    v16h a = frag_a(aRow + kt * 32, lane);
#pragma unroll
    for (int ct = 0; ct < 4; ++ct) {
      const _Float16* bp = B1 + (size_t)(colBase + ct * 16 + jn) * FF + kt * 32;
      acc[ct] = wmma16(a, frag_b(bp, lane), acc[ct]);
    }
  }
  if (A2) {
    const _Float16* aRow2 = A2 + (size_t)(rowBase + jn) * FF;
    for (int kt = 0; kt < KT; ++kt) {
      v16h a = frag_a(aRow2 + kt * 32, lane);
#pragma unroll
      for (int ct = 0; ct < 4; ++ct) {
        const _Float16* bp = B2 + (size_t)(colBase + ct * 16 + jn) * FF + kt * 32;
        acc[ct] = wmma16(a, frag_b(bp, lane), acc[ct]);
      }
    }
  }
#pragma unroll
  for (int ct = 0; ct < 4; ++ct) {
    int j = colBase + ct * 16 + jn;
    float bv = 0.f;
    if (bias1) bv += bias1[j];
    if (bias2) bv += bias2[j];
#pragma unroll
    for (int v = 0; v < 8; ++v)
      C[(size_t)(rowBase + v + hi8) * ncols + j] = (_Float16)(acc[ct][v] + bv);
  }
}

// ---- LSTM recurrence over the mailbox --------------------------------------
// P  : [NN, 2048] f16 = x@Wih^T + bih + bhh (gathered via src each step).
// Whh: [2048, 512] f16.  Block owns 32 nodes (2 row-tiles, so each loaded B
// fragment feeds 2 WMMAs -> halves the dominant Whh L2 traffic).
// RT=2 is the register sweet spot: ~190 VGPRs, keeps kt/ct loops fully
// unrolled and spill-free (RT=4 forced the compiler to roll the loops and
// spill c-state, verified via static histogram regression).
// Wave w owns output cols [w*64, w*64+64).  h double-buffered fp16 in LDS;
// c in registers. One workgroup barrier per step. Tail tile clamps gathers
// and guards global writes (NN = 625*16 row-tiles, grid = 313 blocks of 2).
#define RT 2
__global__ __launch_bounds__(256) void k_lstm(
    const _Float16* __restrict__ P,
    const _Float16* __restrict__ Whh,
    const int* __restrict__ src,
    _Float16* __restrict__ outH16,   // stage-1 result (f16) or nullptr
    float* __restrict__ outF32,      // final result (f32) or nullptr
    int writeF32)
{
  __shared__ _Float16 hb[2][RT * 16 * FF];   // 2 x 32KB double buffer

  const int tid = threadIdx.x, lane = tid & 31, wave = tid >> 5;
  const int nodeBase = blockIdx.x * (RT * 16);
  const int colBase = wave << 6;
  const int jn = lane & 15;
  const int hi8 = (lane >> 4) << 3;

  bool tileValid[RT];
#pragma unroll
  for (int rt = 0; rt < RT; ++rt) tileValid[rt] = (nodeBase + rt * 16) < NN;

  for (int i = tid; i < RT * 16 * FF; i += 256) hb[0][i] = (_Float16)0.f;
  __syncthreads();

  float cst[4][RT][8];
#pragma unroll
  for (int ct = 0; ct < 4; ++ct)
#pragma unroll
    for (int rt = 0; rt < RT; ++rt)
#pragma unroll
      for (int v = 0; v < 8; ++v) cst[ct][rt][v] = 0.f;

  int cur = 0;
  for (int t = 0; t < DD; ++t) {
    int sIdx[RT][8];
#pragma unroll
    for (int rt = 0; rt < RT; ++rt)
#pragma unroll
      for (int v = 0; v < 8; ++v) {
        int n = nodeBase + rt * 16 + v + hi8;
        if (n >= NN) n = NN - 1;                 // clamp tail gathers
        sIdx[rt][v] = src[(size_t)n * DD + t];
      }

#pragma unroll
    for (int ct = 0; ct < 4; ++ct) {
      v8f acc[4][RT];
#pragma unroll
      for (int g = 0; g < 4; ++g)
#pragma unroll
        for (int rt = 0; rt < RT; ++rt)
          acc[g][rt] = (v8f){0.f,0.f,0.f,0.f,0.f,0.f,0.f,0.f};

      if (t) {  // h == 0 at t==0; skip the dead GEMM
        for (int kt = 0; kt < KT; ++kt) {
          v16h a[RT];
#pragma unroll
          for (int rt = 0; rt < RT; ++rt)
            a[rt] = frag_a(&hb[cur][(rt * 16 + jn) * FF + kt * 32], lane);
#pragma unroll
          for (int g = 0; g < 4; ++g) {
            const _Float16* bp =
                Whh + (size_t)(g * FF + colBase + ct * 16 + jn) * FF + kt * 32;
            v16h b = frag_b(bp, lane);           // loaded once, used RT times
#pragma unroll
            for (int rt = 0; rt < RT; ++rt)
              acc[g][rt] = wmma16(a[rt], b, acc[g][rt]);
          }
        }
      }
      // gate math + state update for this 16-col tile
      int j = colBase + ct * 16 + jn;
#pragma unroll
      for (int rt = 0; rt < RT; ++rt) {
#pragma unroll
        for (int v = 0; v < 8; ++v) {
          const _Float16* Pr = P + (size_t)sIdx[rt][v] * G4;
          float gi = acc[0][rt][v] + (float)Pr[j];
          float gf = acc[1][rt][v] + (float)Pr[FF + j];
          float gg = acc[2][rt][v] + (float)Pr[2 * FF + j];
          float go = acc[3][rt][v] + (float)Pr[3 * FF + j];
          float c = sigm(gf) * cst[ct][rt][v] + sigm(gi) * tanhf(gg);
          float h = sigm(go) * tanhf(c);
          cst[ct][rt][v] = c;
          int m = v + hi8;
          hb[cur ^ 1][(rt * 16 + m) * FF + j] = (_Float16)h;
          if (t == DD - 1 && tileValid[rt]) {
            size_t o = (size_t)(nodeBase + rt * 16 + m) * FF + j;
            if (writeF32) outF32[o] = h;
            else          outH16[o] = (_Float16)h;
          }
        }
      }
    }
    __syncthreads();   // writes to hb[cur^1] visible before next step's reads
    cur ^= 1;
  }
}

// ---- host-side orchestration ------------------------------------------------
extern "C" void kernel_launch(void* const* d_in, const int* in_sizes, int n_in,
                              void* d_out, int out_size, void* d_ws, size_t ws_size,
                              hipStream_t stream) {
  (void)in_sizes; (void)n_in; (void)out_size; (void)ws_size;

  const float* x      = (const float*)d_in[0];
  const int*   src    = (const int*)  d_in[1];
  const float* Wih1   = (const float*)d_in[2];
  const float* Whh1   = (const float*)d_in[3];
  const float* bih1   = (const float*)d_in[4];
  const float* bhh1   = (const float*)d_in[5];
  const float* Wself  = (const float*)d_in[6];
  const float* Wneigh = (const float*)d_in[7];
  const float* bvec   = (const float*)d_in[8];
  const float* Wih2   = (const float*)d_in[9];
  const float* Whh2   = (const float*)d_in[10];
  const float* bih2   = (const float*)d_in[11];
  const float* bhh2   = (const float*)d_in[12];
  float* out = (float*)d_out;

  // workspace carve-up (fp16 staging; ~80 MB total)
  char* ws = (char*)d_ws;
  size_t off = 0;
  auto take = [&](size_t bytes) -> void* {
    void* p = ws + off;
    off += (bytes + 255) & ~(size_t)255;
    return p;
  };
  _Float16* x16     = (_Float16*)take((size_t)NN * FF * 2);
  _Float16* wih1_16 = (_Float16*)take((size_t)G4 * FF * 2);
  _Float16* whh1_16 = (_Float16*)take((size_t)G4 * FF * 2);
  _Float16* wih2_16 = (_Float16*)take((size_t)G4 * FF * 2);
  _Float16* whh2_16 = (_Float16*)take((size_t)G4 * FF * 2);
  _Float16* wselfT  = (_Float16*)take((size_t)FF * FF * 2);
  _Float16* wneighT = (_Float16*)take((size_t)FF * FF * 2);
  _Float16* P16     = (_Float16*)take((size_t)NN * G4 * 2);  // reused for P1, P2
  _Float16* hn16    = (_Float16*)take((size_t)NN * FF * 2);  // stage-1 LSTM out
  _Float16* h16     = (_Float16*)take((size_t)NN * FF * 2);  // SAGE mixed features

  const int lstmBlocks = (NN + RT * 16 - 1) / (RT * 16);     // 313

  // 1) precision staging
  k_cvt<<<dim3(2048), 256, 0, stream>>>(x,    x16,     NN * FF);
  k_cvt<<<dim3(1024), 256, 0, stream>>>(Wih1, wih1_16, G4 * FF);
  k_cvt<<<dim3(1024), 256, 0, stream>>>(Whh1, whh1_16, G4 * FF);
  k_cvt<<<dim3(1024), 256, 0, stream>>>(Wih2, wih2_16, G4 * FF);
  k_cvt<<<dim3(1024), 256, 0, stream>>>(Whh2, whh2_16, G4 * FF);
  k_tr <<<dim3(512),  256, 0, stream>>>(Wself,  wselfT);
  k_tr <<<dim3(512),  256, 0, stream>>>(Wneigh, wneighT);

  // 2) P1 = x @ Wih1^T + bih1 + bhh1   [NN, 2048]
  k_gemm<<<dim3(NN / 16, 4), 256, 0, stream>>>(
      x16, wih1_16, nullptr, nullptr, bih1, bhh1, P16, G4);

  // 3) stage-1 LSTM over gathered P1 -> h_neigh (f16)
  k_lstm<<<dim3(lstmBlocks), 256, 0, stream>>>(
      P16, whh1_16, src, hn16, nullptr, 0);

  // 4) h = x @ W_self + h_neigh @ W_neigh + b   [NN, 512]
  k_gemm<<<dim3(NN / 16, 1), 256, 0, stream>>>(
      x16, wselfT, hn16, wneighT, bvec, nullptr, h16, FF);

  // 5) P2 = h @ Wih2^T + bih2 + bhh2   [NN, 2048]
  k_gemm<<<dim3(NN / 16, 4), 256, 0, stream>>>(
      h16, wih2_16, nullptr, nullptr, bih2, bhh2, P16, G4);

  // 6) stage-2 LSTM over gathered P2 -> out (f32)
  k_lstm<<<dim3(lstmBlocks), 256, 0, stream>>>(
      P16, whh2_16, src, nullptr, out, 1);
}
